// TreeDecoder_7516192768462
// MI455X (gfx1250) — compile-verified, matrix operand order checked by reading, not run on hardware
//
#include <hip/hip_runtime.h>
#include <stdint.h>

// ---------------------------------------------------------------------------
// TreeDecoder for MI455X (gfx1250), bf16 WMMA (v_wmma_f32_16x16x32_bf16).
// B=16 batch == WMMA M=16, so every GEMM is a single row-tile of WMMA waves.
// All weights converted once per launch to bf16 in d_ws (working set ~67MB,
// fits MI455X 192MB L2 -> after first touch everything is L2-resident).
// ---------------------------------------------------------------------------

#define B_  16
#define T_  64
#define LX_ 64
#define D_  512
#define V_  32128

typedef __attribute__((ext_vector_type(16))) __bf16 v16bf;
typedef __attribute__((ext_vector_type(8)))  float  v8f;

__device__ __forceinline__ uint16_t f2bf(float f) {
  union { float f; uint32_t u; } x; x.f = f;
  uint32_t u = x.u;
  return (uint16_t)((u + 0x7FFFu + ((u >> 16) & 1u)) >> 16);  // RNE truncate
}
__device__ __forceinline__ float sigm(float x) { return 1.0f / (1.0f + __expf(-x)); }

// ---------------------------------------------------------------------------
// Generic M=16 WMMA GEMM:  C[16 x N] (f32, row stride ldc) =
//    A[16 x K] (bf16, row stride lda) @ W[N x K]^T (bf16, contiguous) + bias
// One wave per 16-wide N tile; K stepped by 32 with 16x16x32 bf16 WMMA.
// blockDim must be 128 (4 waves); grid = N/64 (all our N are multiples of 64).
// ---------------------------------------------------------------------------
__global__ void wmma_gemm16(const uint16_t* __restrict__ A, int lda,
                            const uint16_t* __restrict__ W,
                            const float* __restrict__ bias,
                            float* __restrict__ C, long long ldc,
                            int N, int K)
{
  int wave = threadIdx.x >> 5;
  int lane = threadIdx.x & 31;
  int tile = blockIdx.x * (blockDim.x >> 5) + wave;   // wave-uniform
  if (tile * 16 >= N) return;                          // uniform exit keeps EXEC full
  int n0 = tile * 16;
  int m  = lane & 15;
  int hi = lane >> 4;

  // A fragment: lane(m,hi) holds K = [hi*8 .. hi*8+7] and [16+hi*8 .. +7]
  const uint16_t* arow = A + (size_t)m * lda + hi * 8;
  // B fragment: same contiguous pattern over rows of W (W is B^T row-major)
  const uint16_t* wrow = W + (size_t)(n0 + m) * K + hi * 8;

  v8f acc = {};
  for (int k0 = 0; k0 < K; k0 += 32) {
    union { uint4 q[2]; v16bf v; } af, bfv;
    af.q[0]  = *(const uint4*)(arow + k0);
    af.q[1]  = *(const uint4*)(arow + k0 + 16);
    bfv.q[0] = *(const uint4*)(wrow + k0);
    bfv.q[1] = *(const uint4*)(wrow + k0 + 16);
    acc = __builtin_amdgcn_wmma_f32_16x16x32_bf16(
        false, af.v, false, bfv.v, (short)0, acc, false, false);
  }

  int n = n0 + (lane & 15);
  float bval = bias ? bias[n] : 0.0f;
#pragma unroll
  for (int r = 0; r < 8; ++r) {
    int mm = r + hi * 8;                    // C/D layout: VGPR r -> row hi*8+r
    C[(long long)mm * ldc + n] = acc[r] + bval;
  }
}

// ---------------------------------------------------------------------------
// Per-step pointwise LSTM cell (+optional write-mask blend for the word cell).
// gates: [16 x 2048] f32 (i|f|g|o).  Updates h,c in place; emits bf16 h2 into
// the readout `inp` slot, the attention query buffer, optionally the r-LSTM
// input slot, and optionally states[t+1].
// ---------------------------------------------------------------------------
__global__ void lstm_pointwise(const float* __restrict__ gates,
                               float* __restrict__ h, float* __restrict__ c,
                               const int* __restrict__ y_train, int t, int use_mask,
                               uint16_t* __restrict__ inp_bf, int inp_off,
                               uint16_t* __restrict__ q_bf,
                               uint16_t* __restrict__ rx_slot,      // or null
                               float* __restrict__ states_out)      // or null
{
  int tid = blockIdx.x * blockDim.x + threadIdx.x;
  if (tid >= B_ * D_) return;
  int b = tid >> 9, d = tid & (D_ - 1);
  const float* g = gates + b * 2048;
  float ig = g[d], fg = g[D_ + d], gg = g[2 * D_ + d], og = g[3 * D_ + d];
  float c_old = c[tid], h_old = h[tid];
  float c2 = sigm(fg) * c_old + sigm(ig) * tanhf(gg);
  float h2 = sigm(og) * tanhf(c2);
  if (use_mask) {
    float wmf = (float)y_train[(b * T_ + t) * 3 + 2];
    h2 = h2 * wmf + h_old * (1.0f - wmf);
    c2 = c2 * wmf + c_old * (1.0f - wmf);
  }
  h[tid] = h2; c[tid] = c2;
  uint16_t hb = f2bf(h2);
  inp_bf[b * 3072 + inp_off + d] = hb;
  q_bf[tid] = hb;
  if (rx_slot)    rx_slot[b * 3072 + 2048 + d] = hb;
  if (states_out) states_out[tid] = h2;
}

// ---------------------------------------------------------------------------
// Build bf16 LSTM inputs for step t.
//  w_x[16x2560] = [y_emb | parent_state | wf | wh_old]
//  r_x[16x3072] = [y_emb | parent_state | rf | (wh2: filled later) | rh_old]
// ---------------------------------------------------------------------------
__global__ void build_inputs(const uint16_t* __restrict__ trgemb,
                             const int* __restrict__ y_train,
                             const float* __restrict__ states,
                             const float* __restrict__ rf, const float* __restrict__ wf,
                             const float* __restrict__ rh, const float* __restrict__ wh,
                             uint16_t* __restrict__ w_x, uint16_t* __restrict__ r_x, int t)
{
  int tid = blockIdx.x * blockDim.x + threadIdx.x;
  if (tid >= B_ * 3072) return;
  int b = tid / 3072, j = tid - b * 3072;
  int parent = y_train[(b * T_ + t) * 3 + 1];
  const float* pstate = states + (size_t)parent * B_ * D_ + b * D_;

  // r_x
  if (j < 512)       r_x[b*3072+j] = trgemb[(b * T_ + t) * D_ + j];
  else if (j < 1024) r_x[b*3072+j] = f2bf(pstate[j - 512]);
  else if (j < 2048) r_x[b*3072+j] = f2bf(rf[b * 1024 + (j - 1024)]);
  else if (j >= 2560) r_x[b*3072+j] = f2bf(rh[b * D_ + (j - 2560)]);
  // (j in [2048,2560) is wh2, written by lstm_pointwise of the word cell)

  // w_x
  if (j < 2560) {
    uint16_t wv;
    if (j < 512)       wv = trgemb[(b * T_ + t) * D_ + j];
    else if (j < 1024) wv = f2bf(pstate[j - 512]);
    else if (j < 2048) wv = f2bf(wf[b * 1024 + (j - 1024)]);
    else               wv = f2bf(wh[b * D_ + (j - 2048)]);
    w_x[b * 2560 + j] = wv;
  }
}

// ---------------------------------------------------------------------------
// Attention scores for both attentions: s[b][l] = sum_d tanh(k+proj)*wa + ba.
// One wave per (attn, b, l) row; lane-strided dot + shuffle reduction.
// ---------------------------------------------------------------------------
__global__ void attn_scores(const float* __restrict__ x_enc_k,
                            const unsigned char* __restrict__ x_mask,
                            const float* __restrict__ proj_r, const float* __restrict__ proj_w,
                            const float* __restrict__ r_wa, const float* __restrict__ r_ba,
                            const float* __restrict__ w_wa, const float* __restrict__ w_ba,
                            float* __restrict__ s_r, float* __restrict__ s_w)
{
  int wave = threadIdx.x >> 5, lane = threadIdx.x & 31;
  int gw = blockIdx.x * (blockDim.x >> 5) + wave;
  if (gw >= 2 * B_ * LX_) return;
  int a = gw >> 10, rem = gw & 1023, b = rem >> 6, l = rem & 63;
  const float* proj = (a ? proj_w : proj_r) + b * D_;
  const float* wa   = a ? w_wa : r_wa;
  const float* krow = x_enc_k + (size_t)(b * LX_ + l) * D_;
  float sum = 0.0f;
#pragma unroll 4
  for (int i = 0; i < D_ / 32; ++i) {
    int d = lane + 32 * i;
    sum += tanhf(krow[d] + proj[d]) * wa[d];
  }
  for (int off = 16; off; off >>= 1) sum += __shfl_xor(sum, off, 32);
  if (lane == 0) {
    float s = sum + (a ? w_ba : r_ba)[0];
    if (x_mask[b * LX_ + l]) s = -__builtin_inff();
    (a ? s_w : s_r)[b * LX_ + l] = s;
  }
}

// ---------------------------------------------------------------------------
// Softmax over LX + context  ctx[b] = sum_l a[l] * x_enc[b][l][:1024].
// One block per (attn, b). Writes f32 feed (rf/wf) and bf16 readout slot.
// ---------------------------------------------------------------------------
__global__ void attn_ctx(const float* __restrict__ x_enc,
                         const float* __restrict__ s_r, const float* __restrict__ s_w,
                         float* __restrict__ rf, float* __restrict__ wf,
                         uint16_t* __restrict__ inp_bf)
{
  __shared__ float sh[LX_];
  __shared__ float swt[LX_];
  int a = blockIdx.x >> 4, b = blockIdx.x & 15;
  int tid = threadIdx.x;
  const float* s = (a ? s_w : s_r) + b * LX_;
  if (tid < LX_) sh[tid] = s[tid];
  __syncthreads();
  float mx = -__builtin_inff();
  for (int l = 0; l < LX_; ++l) mx = fmaxf(mx, sh[l]);
  float den = 0.0f;
  for (int l = 0; l < LX_; ++l) den += __expf(sh[l] - mx);
  if (tid < LX_) swt[tid] = __expf(sh[tid] - mx) / den;
  __syncthreads();
  float* feed = (a ? wf : rf) + b * 1024;
  uint16_t* slot = inp_bf + b * 3072 + (a ? 2048 : 512);
  for (int d = tid; d < 1024; d += blockDim.x) {
    float acc = 0.0f;
    for (int l = 0; l < LX_; ++l)
      acc += swt[l] * x_enc[(size_t)(b * LX_ + l) * 1024 + d];
    feed[d] = acc;
    slot[d] = f2bf(acc);
  }
}

// score = (wmask ? tanh(pre_word) : tanh(pre_rule)) @ readout^T  -- matmul is
// linear per row, so select the 16x512 pre-vector BEFORE the big V matmul.
__global__ void select_tanh(const float* __restrict__ pre_rule,
                            const float* __restrict__ pre_word,
                            const int* __restrict__ y_train, int t,
                            uint16_t* __restrict__ pre_sel)
{
  int tid = blockIdx.x * blockDim.x + threadIdx.x;
  if (tid >= B_ * D_) return;
  int b = tid >> 9;
  int wm = y_train[(b * T_ + t) * 3 + 2];
  float v = wm ? pre_word[tid] : pre_rule[tid];
  pre_sel[tid] = f2bf(tanhf(v));
}

// ------------------------------ precompute ---------------------------------
__global__ void cvt_bf16(const float* __restrict__ src, uint16_t* __restrict__ dst, int n) {
  int tid = blockIdx.x * blockDim.x + threadIdx.x;
  if (tid < n) dst[tid] = f2bf(src[tid]);
}
// dst[N=2048 x Kc] = [Wih (2048 x Kih) | Whh (2048 x 512)] in bf16
__global__ void concat_w(const float* __restrict__ Wih, int Kih,
                         const float* __restrict__ Whh,
                         uint16_t* __restrict__ dst, int Kc, int n_total) {
  int tid = blockIdx.x * blockDim.x + threadIdx.x;
  if (tid >= n_total) return;
  int n = tid / Kc, k = tid - n * Kc;
  float v = (k < Kih) ? Wih[(size_t)n * Kih + k] : Whh[(size_t)n * D_ + (k - Kih)];
  dst[tid] = f2bf(v);
}
__global__ void bias_sum(const float* rbi, const float* rbh,
                         const float* wbi, const float* wbh,
                         float* br, float* bw) {
  int tid = blockIdx.x * blockDim.x + threadIdx.x;
  if (tid < 2048) { br[tid] = rbi[tid] + rbh[tid]; bw[tid] = wbi[tid] + wbh[tid]; }
}
__global__ void gather_emb(const float* __restrict__ emb, const int* __restrict__ y_train,
                           uint16_t* __restrict__ dst) {
  int tid = blockIdx.x * blockDim.x + threadIdx.x;
  if (tid >= B_ * T_ * D_) return;
  int bt = tid >> 9, d = tid & (D_ - 1);
  int tok = y_train[bt * 3];
  dst[tid] = f2bf(emb[(size_t)tok * D_ + d]);
}
__global__ void zero_f32(float* p, int n) {
  int tid = blockIdx.x * blockDim.x + threadIdx.x;
  if (tid < n) p[tid] = 0.0f;
}
__global__ void init_hc(const float* h0, const float* c0,
                        float* rh, float* rc, float* wh, float* wc) {
  int tid = blockIdx.x * blockDim.x + threadIdx.x;
  if (tid < B_ * D_) { rh[tid] = h0[tid]; wh[tid] = h0[tid];
                       rc[tid] = c0[tid]; wc[tid] = c0[tid]; }
}

// ---------------------------------------------------------------------------
extern "C" void kernel_launch(void* const* d_in, const int* in_sizes, int n_in,
                              void* d_out, int out_size, void* d_ws, size_t ws_size,
                              hipStream_t stream) {
  (void)in_sizes; (void)n_in; (void)out_size; (void)ws_size;
  const float* x_enc      = (const float*)d_in[0];
  const float* x_enc_k    = (const float*)d_in[1];
  const float* dec_init_h = (const float*)d_in[2];
  const float* dec_init_c = (const float*)d_in[3];
  const int*   y_train    = (const int*)d_in[4];
  const unsigned char* x_mask = (const unsigned char*)d_in[5];
  const float* emb        = (const float*)d_in[6];
  const float* ra_W  = (const float*)d_in[7];
  const float* ra_b  = (const float*)d_in[8];
  const float* ra_wa = (const float*)d_in[9];
  const float* ra_ba = (const float*)d_in[10];
  const float* wa_W  = (const float*)d_in[11];
  const float* wa_b  = (const float*)d_in[12];
  const float* wa_wa = (const float*)d_in[13];
  const float* wa_ba = (const float*)d_in[14];
  const float* rule_W    = (const float*)d_in[15];
  const float* word_W    = (const float*)d_in[16];
  const float* readout_W = (const float*)d_in[17];
  const float* r_Wih = (const float*)d_in[18];
  const float* r_Whh = (const float*)d_in[19];
  const float* r_bih = (const float*)d_in[20];
  const float* r_bhh = (const float*)d_in[21];
  const float* w_Wih = (const float*)d_in[22];
  const float* w_Whh = (const float*)d_in[23];
  const float* w_bih = (const float*)d_in[24];
  const float* w_bhh = (const float*)d_in[25];
  float* out = (float*)d_out;

  // ---- workspace bump allocator (256B aligned) ----
  char* base = (char*)d_ws; size_t off = 0;
  auto alloc = [&](size_t bytes) -> void* {
    void* p = base + off; off = (off + bytes + 255) & ~(size_t)255; return p;
  };
  uint16_t* readout_bf = (uint16_t*)alloc((size_t)V_ * D_ * 2);
  uint16_t* rcat_bf    = (uint16_t*)alloc((size_t)2048 * 3072 * 2);
  uint16_t* wcat_bf    = (uint16_t*)alloc((size_t)2048 * 2560 * 2);
  uint16_t* rule_bf    = (uint16_t*)alloc((size_t)D_ * 3072 * 2);
  uint16_t* word_bf    = (uint16_t*)alloc((size_t)D_ * 3072 * 2);
  uint16_t* raW_bf     = (uint16_t*)alloc((size_t)D_ * D_ * 2);
  uint16_t* waW_bf     = (uint16_t*)alloc((size_t)D_ * D_ * 2);
  uint16_t* trgemb_bf  = (uint16_t*)alloc((size_t)B_ * T_ * D_ * 2);
  float* bsum_r  = (float*)alloc(2048 * 4);
  float* bsum_w  = (float*)alloc(2048 * 4);
  float* states  = (float*)alloc((size_t)(T_ + 1) * B_ * D_ * 4);
  float* rh = (float*)alloc(B_ * D_ * 4);
  float* rc = (float*)alloc(B_ * D_ * 4);
  float* wh = (float*)alloc(B_ * D_ * 4);
  float* wc = (float*)alloc(B_ * D_ * 4);
  float* rfeed = (float*)alloc(B_ * 1024 * 4);
  float* wfeed = (float*)alloc(B_ * 1024 * 4);
  float* gates_r = (float*)alloc(B_ * 2048 * 4);
  float* gates_w = (float*)alloc(B_ * 2048 * 4);
  float* proj_r = (float*)alloc(B_ * D_ * 4);
  float* proj_w = (float*)alloc(B_ * D_ * 4);
  float* s_r = (float*)alloc(B_ * LX_ * 4);
  float* s_w = (float*)alloc(B_ * LX_ * 4);
  float* pre_rule = (float*)alloc(B_ * D_ * 4);
  float* pre_word = (float*)alloc(B_ * D_ * 4);
  uint16_t* w_x     = (uint16_t*)alloc(B_ * 2560 * 2);
  uint16_t* r_x     = (uint16_t*)alloc(B_ * 3072 * 2);
  uint16_t* q_r     = (uint16_t*)alloc(B_ * D_ * 2);
  uint16_t* q_w     = (uint16_t*)alloc(B_ * D_ * 2);
  uint16_t* inp_bf  = (uint16_t*)alloc(B_ * 3072 * 2);
  uint16_t* pre_sel = (uint16_t*)alloc(B_ * D_ * 2);

  auto cdiv = [](int a, int b) { return (a + b - 1) / b; };

  // ---- one-time precompute (re-done every call: deterministic) ----
  cvt_bf16<<<cdiv(V_ * D_, 256), 256, 0, stream>>>(readout_W, readout_bf, V_ * D_);
  cvt_bf16<<<cdiv(D_ * 3072, 256), 256, 0, stream>>>(rule_W, rule_bf, D_ * 3072);
  cvt_bf16<<<cdiv(D_ * 3072, 256), 256, 0, stream>>>(word_W, word_bf, D_ * 3072);
  cvt_bf16<<<cdiv(D_ * D_, 256), 256, 0, stream>>>(ra_W, raW_bf, D_ * D_);
  cvt_bf16<<<cdiv(D_ * D_, 256), 256, 0, stream>>>(wa_W, waW_bf, D_ * D_);
  concat_w<<<cdiv(2048 * 3072, 256), 256, 0, stream>>>(r_Wih, 2560, r_Whh, rcat_bf, 3072, 2048 * 3072);
  concat_w<<<cdiv(2048 * 2560, 256), 256, 0, stream>>>(w_Wih, 2048, w_Whh, wcat_bf, 2560, 2048 * 2560);
  bias_sum<<<cdiv(2048, 256), 256, 0, stream>>>(r_bih, r_bhh, w_bih, w_bhh, bsum_r, bsum_w);
  gather_emb<<<cdiv(B_ * T_ * D_, 256), 256, 0, stream>>>(emb, y_train, trgemb_bf);
  zero_f32<<<cdiv((T_ + 1) * B_ * D_, 256), 256, 0, stream>>>(states, (T_ + 1) * B_ * D_);
  zero_f32<<<cdiv(B_ * 1024, 256), 256, 0, stream>>>(rfeed, B_ * 1024);
  zero_f32<<<cdiv(B_ * 1024, 256), 256, 0, stream>>>(wfeed, B_ * 1024);
  init_hc<<<cdiv(B_ * D_, 256), 256, 0, stream>>>(dec_init_h, dec_init_c, rh, rc, wh, wc);

  // ---- sequential scan over T steps ----
  for (int t = 0; t < T_; ++t) {
    build_inputs<<<cdiv(B_ * 3072, 256), 256, 0, stream>>>(
        trgemb_bf, y_train, states, rfeed, wfeed, rh, wh, w_x, r_x, t);

    // word LSTM gates: [16x2560] @ [2048x2560]^T
    wmma_gemm16<<<2048 / 64, 128, 0, stream>>>(w_x, 2560, wcat_bf, bsum_w,
                                               gates_w, 2048, 2048, 2560);
    lstm_pointwise<<<cdiv(B_ * D_, 256), 256, 0, stream>>>(
        gates_w, wh, wc, y_train, t, /*use_mask=*/1,
        inp_bf, /*inp_off=*/1536, q_w, /*rx_slot=*/r_x, /*states_out=*/nullptr);

    // rule LSTM gates: [16x3072] @ [2048x3072]^T
    wmma_gemm16<<<2048 / 64, 128, 0, stream>>>(r_x, 3072, rcat_bf, bsum_r,
                                               gates_r, 2048, 2048, 3072);
    lstm_pointwise<<<cdiv(B_ * D_, 256), 256, 0, stream>>>(
        gates_r, rh, rc, y_train, t, /*use_mask=*/0,
        inp_bf, /*inp_off=*/0, q_r, /*rx_slot=*/nullptr,
        /*states_out=*/states + (size_t)(t + 1) * B_ * D_);

    // attention target projections: [16x512] @ [512x512]^T + b
    wmma_gemm16<<<512 / 64, 128, 0, stream>>>(q_r, 512, raW_bf, ra_b,
                                              proj_r, 512, 512, 512);
    wmma_gemm16<<<512 / 64, 128, 0, stream>>>(q_w, 512, waW_bf, wa_b,
                                              proj_w, 512, 512, 512);

    attn_scores<<<(2 * B_ * LX_) / 8, 256, 0, stream>>>(
        x_enc_k, x_mask, proj_r, proj_w, ra_wa, ra_ba, wa_wa, wa_ba, s_r, s_w);
    attn_ctx<<<2 * B_, 256, 0, stream>>>(x_enc, s_r, s_w, rfeed, wfeed, inp_bf);

    // readout pre-activations: [16x3072] @ [512x3072]^T
    wmma_gemm16<<<512 / 64, 128, 0, stream>>>(inp_bf, 3072, rule_bf, nullptr,
                                              pre_rule, 512, 512, 3072);
    wmma_gemm16<<<512 / 64, 128, 0, stream>>>(inp_bf, 3072, word_bf, nullptr,
                                              pre_word, 512, 512, 3072);
    select_tanh<<<cdiv(B_ * D_, 256), 256, 0, stream>>>(pre_rule, pre_word,
                                                        y_train, t, pre_sel);

    // vocab logits: [16x512] @ [32128x512]^T  -> out[b][t][v]
    wmma_gemm16<<<V_ / 64, 128, 0, stream>>>(
        pre_sel, 512, readout_bf, nullptr,
        out + (size_t)t * V_, (long long)T_ * V_, V_, 512);
  }
}
